// Int8Linear_89653147336852
// MI455X (gfx1250) — 8-bit matrix operand order repaired and checked by reading, compiled, not run on hardware
//
#include <hip/hip_runtime.h>
#include <hip/hip_fp16.h>

typedef __attribute__((ext_vector_type(8))) int v8i;
typedef __attribute__((ext_vector_type(4))) int v4i;
typedef __attribute__((ext_vector_type(2))) int v2i;

#define BLK_M 128
#define BLK_N 128
#define TILE_K 64
#define LDSB_STRIDE 80

constexpr int GM = 16384;
constexpr int GN = 2048;
constexpr int GK = 2048;

__global__ __launch_bounds__(256, 2)
void int8_gemm_wmma_kernel(const signed char* __restrict__ A,
                           const signed char* __restrict__ B,
                           const __half*     __restrict__ bias,
                           float*            __restrict__ C)
{
    __shared__ signed char ldsB[2][BLK_N * LDSB_STRIDE];

    const int tid   = threadIdx.x;
    const int lane  = tid & 31;
    const int wave  = tid >> 5;
    const int waveM = wave >> 1;
    const int waveN = wave & 1;

    const int blockN0 = blockIdx.x * BLK_N;
    const int blockM0 = blockIdx.y * BLK_M;

    const int mrow  = lane & 15;
    const int khalf = (lane >> 4) << 3;
    const int kq    = (lane >> 4) << 4;

    v8i acc[2][4];
    #pragma unroll
    for (int i = 0; i < 2; ++i)
        #pragma unroll
        for (int j = 0; j < 4; ++j)
            #pragma unroll
            for (int r = 0; r < 8; ++r)
                acc[i][j][r] = 0;

    const signed char* Abase0 = A + (size_t)(blockM0 + waveM * 32 +  0 + mrow) * GK;
    const signed char* Abase1 = A + (size_t)(blockM0 + waveM * 32 + 16 + mrow) * GK;

    auto stageB = [&](int kc, int buf) {
        #pragma unroll
        for (int c = 0; c < 8; ++c) {
            int linear = c * 256 + tid;
            int n  = linear & 127;
            int kb = (linear >> 7) << 2;
            const signed char* gp = B + (size_t)(kc + kb) * GN + (blockN0 + n);
            unsigned char b0 = (unsigned char)gp[0 * GN];
            unsigned char b1 = (unsigned char)gp[1 * GN];
            unsigned char b2 = (unsigned char)gp[2 * GN];
            unsigned char b3 = (unsigned char)gp[3 * GN];
            unsigned int pack = (unsigned int)b0
                              | ((unsigned int)b1 << 8)
                              | ((unsigned int)b2 << 16)
                              | ((unsigned int)b3 << 24);
            *(unsigned int*)(&ldsB[buf][n * LDSB_STRIDE + kb]) = pack;
        }
    };

    stageB(0, 0);

    int buf = 0;
    for (int k0 = 0; k0 < GK; k0 += TILE_K) {
        __syncthreads();

        __builtin_prefetch(Abase0 + k0 + 2 * TILE_K, 0, 1);
        __builtin_prefetch(Abase1 + k0 + 2 * TILE_K, 0, 1);

        v8i afrag[2];
        #pragma unroll
        for (int g = 0; g < 4; ++g) {
            v2i t0 = *(const v2i*)(Abase0 + k0 + (g >> 1) * 32 + (g & 1) * 8 + kq);
            v2i t1 = *(const v2i*)(Abase1 + k0 + (g >> 1) * 32 + (g & 1) * 8 + kq);
            afrag[0][2 * g]     = t0[0];
            afrag[0][2 * g + 1] = t0[1];
            afrag[1][2 * g]     = t1[0];
            afrag[1][2 * g + 1] = t1[1];
        }

        if (k0 + TILE_K < GK)
            stageB(k0 + TILE_K, buf ^ 1);

        v8i bfrag[4];
        #pragma unroll
        for (int nt = 0; nt < 4; ++nt) {
            int nloc = waveN * 64 + nt * 16 + (lane & 15);
            const signed char* bp = &ldsB[buf][nloc * LDSB_STRIDE];
            v4i lo = *(const v4i*)(bp + kq);
            v4i hi = *(const v4i*)(bp + 32 + kq);
            #pragma unroll
            for (int r = 0; r < 4; ++r) {
                bfrag[nt][r]     = lo[r];
                bfrag[nt][r + 4] = hi[r];
            }
        }

        #pragma unroll
        for (int mt = 0; mt < 2; ++mt)
            #pragma unroll
            for (int nt = 0; nt < 4; ++nt)
                acc[mt][nt] = __builtin_amdgcn_wmma_i32_16x16x64_iu8(
                     true, afrag[mt],
                     true, bfrag[nt],
                    acc[mt][nt],
                     false,  false);

        buf ^= 1;
    }

    const int rowHalf = (lane >> 4) << 3;
    #pragma unroll
    for (int mt = 0; mt < 2; ++mt) {
        int Mbase = blockM0 + waveM * 32 + mt * 16 + rowHalf;
        #pragma unroll
        for (int nt = 0; nt < 4; ++nt) {
            int col = blockN0 + waveN * 64 + nt * 16 + (lane & 15);
            float bv = __half2float(bias[col]);
            float* cp = C + (size_t)Mbase * GN + col;
            #pragma unroll
            for (int r = 0; r < 8; ++r) {
                cp[(size_t)r * GN] = (float)acc[mt][nt][r] + bv;
            }
        }
    }
}

extern "C" void kernel_launch(void* const* d_in, const int* in_sizes, int n_in,
                              void* d_out, int out_size, void* d_ws, size_t ws_size,
                              hipStream_t stream) {
    const signed char* x    = (const signed char*)d_in[0];
    const signed char* w    = (const signed char*)d_in[1];
    const __half*      bias = (const __half*)d_in[2];
    float*             out  = (float*)d_out;

    dim3 grid(GN / BLK_N, GM / BLK_M);
    int8_gemm_wmma_kernel<<<grid, 256, 0, stream>>>(x, w, bias, out);
}
